// NNAttentionHead_1692217115189
// MI455X (gfx1250) — matrix operand order repaired and checked by reading, not compiled, not run on hardware
//
#include <hip/hip_runtime.h>
#include <hip/hip_bf16.h>
#include <math.h>

// Problem constants
#define BB 2
#define TT 512
#define CC 128
#define C2 256
#define HSZ 64

typedef __attribute__((ext_vector_type(16))) _Float16 v16h;
typedef __attribute__((ext_vector_type(8)))  float    v8f;

// ---------------------------------------------------------------------------
// Stage A1: xk[b,t,c] = x[b,t,:] @ W1[0:C,:]
//           xq[b,t,c] = x[b,t,:] @ W1[C:2C,:] + b1[c]
//           v [b,t,h] = x[b,t,:] @ Wv + bv
// Tiny (42 MFLOP total) — plain VALU.
// ---------------------------------------------------------------------------
__global__ __launch_bounds__(256) void stage_a_proj(
    const float* __restrict__ x, const float* __restrict__ W1,
    const float* __restrict__ b1, const float* __restrict__ Wv,
    const float* __restrict__ bv,
    float* __restrict__ xk, float* __restrict__ xq, float* __restrict__ vbuf) {
  int tid = blockIdx.x * blockDim.x + threadIdx.x;
  const int n1 = BB * TT * CC;
  if (tid < n1) {
    int c = tid % CC;
    int bt = tid / CC;
    const float* xr = x + (size_t)bt * CC;
    float sk = 0.f, sq = 0.f;
#pragma unroll 4
    for (int d = 0; d < CC; ++d) {
      float xv = xr[d];
      sk += xv * W1[d * CC + c];
      sq += xv * W1[(CC + d) * CC + c];
    }
    xk[tid] = sk;
    xq[tid] = sq + b1[c];
  } else {
    int t2 = tid - n1;
    if (t2 < BB * TT * HSZ) {
      int h = t2 % HSZ;
      int bt = t2 / HSZ;
      const float* xr = x + (size_t)bt * CC;
      float s = bv[h];
#pragma unroll 4
      for (int d = 0; d < CC; ++d) s += xr[d] * Wv[d * HSZ + h];
      vbuf[t2] = s;
    }
  }
}

// ---------------------------------------------------------------------------
// Stage A2: pack W1 (f32 [256][128]) into f16 WMMA B-fragment layout.
// Fragment index: ((kk*8+nn)*32 + lane)*16 + h
//   column n = nn*16 + (lane&15)
//   k        = kk*32 + (lane>=16 ? 8 : 0) + (h&7) + (h>>3)*16  (B: lanes 0-15 hold K {0..7,16..23})
// ---------------------------------------------------------------------------
__global__ __launch_bounds__(256) void stage_a_packW1(
    const float* __restrict__ W1, _Float16* __restrict__ packed) {
  int t = blockIdx.x * blockDim.x + threadIdx.x;  // 32768 threads
  int h    = t & 15;
  int lane = (t >> 4) & 31;
  int nn   = (t >> 9) & 7;
  int kk   = (t >> 12) & 7;
  int n = nn * 16 + (lane & 15);
  int k = kk * 32 + ((lane >> 4) ? 8 : 0) + (h & 7) + ((h >> 3) << 4);
  packed[t] = (_Float16)W1[k * CC + n];
}

// ---------------------------------------------------------------------------
// Stage B: per (i, j-block-of-16) causal tile:
//   pde_h[16x128] = pde[i, j0:j0+16, 0:256] @ W1        (f16 WMMA, f32 acc)
//   for b in {0,1}: score = W2 . gelu(pde_h + xk[b,j] + xq[b,i]) (+b2)*scale
// One wave per tile; 64x v_wmma_f32_16x16x32_f16 per tile.
// ---------------------------------------------------------------------------
__global__ __launch_bounds__(256) void stage_b_scores(
    const float* __restrict__ pde, const _Float16* __restrict__ packedW1,
    const float* __restrict__ xk, const float* __restrict__ xq,
    const float* __restrict__ W2, const float* __restrict__ b2,
    float* __restrict__ scores) {
  __shared__ __align__(32) _Float16 ldsW1[32768];  // 64 KB (WGP has 320 KB)

  // Cooperative LDS fill of the pre-packed W1 fragments (amortized over tiles)
  {
    const uint4* src = (const uint4*)packedW1;
    uint4* dst = (uint4*)ldsW1;
    for (int idx = threadIdx.x; idx < 4096; idx += blockDim.x) dst[idx] = src[idx];
  }
  __syncthreads();
  const v16h* ldsB = (const v16h*)ldsW1;

  const int lane = threadIdx.x & 31;
  const int wave = threadIdx.x >> 5;
  const int wgid = blockIdx.x * (blockDim.x >> 5) + wave;
  const int nwaves = gridDim.x * (blockDim.x >> 5);
  const int half = lane >> 4;   // 0 or 1
  const int l15  = lane & 15;
  const float scale = 0.08838834764831845f;  // 128^-0.5
  const float b2v = b2[0];
  const int NTASK = TT * (TT / 16);          // rectangular task space

  // Per-lane W2 values for this lane's output columns
  float w2r[8];
#pragma unroll
  for (int nn = 0; nn < 8; ++nn) w2r[nn] = W2[nn * 16 + l15];

  for (int task = wgid; task < NTASK; task += nwaves) {
    int i  = task >> 5;          // 32 j-blocks per i
    int j0 = (task & 31) << 4;
    if (j0 > i) continue;        // fully above the causal diagonal

    // Prefetch the next tile's pde rows (global_prefetch_b8)
    {
      int nt = task + nwaves;
      if (nt < NTASK) {
        int ni = nt >> 5, nj0 = (nt & 31) << 4;
        if (nj0 <= ni)
          __builtin_prefetch(pde + ((size_t)ni * TT + nj0 + l15) * C2, 0, 1);
      }
    }

    // A-matrix source row for this lane: M = l15  ->  j = j0 + l15
    const float* arow = pde + ((size_t)i * TT + (j0 + l15)) * C2;

    v8f acc[8];
#pragma unroll
    for (int nn = 0; nn < 8; ++nn)
#pragma unroll
      for (int e = 0; e < 8; ++e) acc[nn][e] = 0.f;

#pragma unroll
    for (int kk = 0; kk < 8; ++kk) {
      // ISA A layout (16-bit, 16x32): lane<16 holds K {0..7,16..23},
      // lane>=16 holds K {8..15,24..31} within this k-step.
      int d0 = kk * 32 + half * 8;
      v16h a;
#pragma unroll
      for (int e = 0; e < 8; ++e) a[e]     = (_Float16)arow[d0 + e];
#pragma unroll
      for (int e = 0; e < 8; ++e) a[8 + e] = (_Float16)arow[d0 + 16 + e];

#pragma unroll
      for (int nn = 0; nn < 8; ++nn) {
        v16h bf = ldsB[(kk * 8 + nn) * 32 + lane];
        acc[nn] = __builtin_amdgcn_wmma_f32_16x16x32_f16(
            false, a, false, bf, (short)0, acc[nn], false, false);
      }
    }

    // Epilogue: C/D layout is lane = N (column), VGPR r = M (+8 for lanes>=16)
#pragma unroll
    for (int b = 0; b < BB; ++b) {
      const float* xkb = xk + (size_t)b * TT * CC;
      const float* xqi = xq + ((size_t)b * TT + i) * CC;
      float xqr[8];
#pragma unroll
      for (int nn = 0; nn < 8; ++nn) xqr[nn] = xqi[nn * 16 + l15];

      float part[8];
#pragma unroll
      for (int r = 0; r < 8; ++r) {
        int j = j0 + r + half * 8;
        const float* xkrow = xkb + (size_t)j * CC;
        float s = 0.f;
#pragma unroll
        for (int nn = 0; nn < 8; ++nn) {
          float pre = acc[nn][r] + xkrow[nn * 16 + l15] + xqr[nn];
          float g = 0.5f * pre * (1.0f + erff(pre * 0.70710678118654752f));
          s += g * w2r[nn];
        }
        part[r] = s;
      }

      // Reduce each row's partial across its 16-lane group; lanes 0-7 end up
      // with rows m=0..7, lanes 16-23 with rows m=8..15.
      float mine = 0.f;
#pragma unroll
      for (int r = 0; r < 8; ++r) {
        float s = part[r];
        s += __shfl_xor(s, 1, 32);
        s += __shfl_xor(s, 2, 32);
        s += __shfl_xor(s, 4, 32);
        s += __shfl_xor(s, 8, 32);
        if (l15 == r) mine = s;
      }
      if (l15 < 8) {
        int m = l15 + half * 8;
        int j = j0 + m;
        if (j <= i) {
          scores[((size_t)b * TT + i) * TT + j] = (mine + b2v) * scale;
        }
      }
    }
  }
}

// ---------------------------------------------------------------------------
// Stage C: per (b,i) row: causal softmax over j<=i, then out = wei @ v.
// One 256-thread block per row (1024 rows).
// ---------------------------------------------------------------------------
__global__ __launch_bounds__(256) void stage_c_softmax_av(
    const float* __restrict__ scores, const float* __restrict__ vbuf,
    float* __restrict__ out) {
  __shared__ float wei[TT];
  __shared__ float red[256];
  const int bi = blockIdx.x;       // b*T + i
  const int i = bi & (TT - 1);
  const int b = bi >> 9;
  const int len = i + 1;
  const int tid = threadIdx.x;
  const float* srow = scores + (size_t)bi * TT;

  // Row max
  float mx = -INFINITY;
  for (int j = tid; j < len; j += blockDim.x) mx = fmaxf(mx, srow[j]);
  red[tid] = mx;
  __syncthreads();
  for (int s = 128; s > 0; s >>= 1) {
    if (tid < s) red[tid] = fmaxf(red[tid], red[tid + s]);
    __syncthreads();
  }
  mx = red[0];
  __syncthreads();

  // exp + sum
  float sum = 0.f;
  for (int j = tid; j < len; j += blockDim.x) {
    float e = __expf(srow[j] - mx);
    wei[j] = e;
    sum += e;
  }
  red[tid] = sum;
  __syncthreads();
  for (int s = 128; s > 0; s >>= 1) {
    if (tid < s) red[tid] += red[tid + s];
    __syncthreads();
  }
  sum = red[0];
  __syncthreads();
  const float inv = 1.0f / sum;

  // out[bi, h] = sum_j wei[j] * v[b, j, h] * inv
  const int h = tid & (HSZ - 1);
  const int chunk = tid >> 6;  // 4 j-chunks
  const float* vb = vbuf + (size_t)b * TT * HSZ;
  float o = 0.f;
  for (int j = chunk; j < len; j += 4) o += wei[j] * vb[j * HSZ + h];
  red[tid] = o;
  __syncthreads();
  if (chunk == 0) {
    o = red[tid] + red[tid + 64] + red[tid + 128] + red[tid + 192];
    out[(size_t)bi * HSZ + h] = o * inv;
  }
}

// ---------------------------------------------------------------------------
// kernel_launch
// d_in order: x, pos_emb(unused), pos_dist_emb, W1, b1, W2, b2, Wv, bv
// ---------------------------------------------------------------------------
extern "C" void kernel_launch(void* const* d_in, const int* in_sizes, int n_in,
                              void* d_out, int out_size, void* d_ws, size_t ws_size,
                              hipStream_t stream) {
  const float* x   = (const float*)d_in[0];
  const float* pde = (const float*)d_in[2];
  const float* W1  = (const float*)d_in[3];
  const float* b1  = (const float*)d_in[4];
  const float* W2  = (const float*)d_in[5];
  const float* b2  = (const float*)d_in[6];
  const float* Wv  = (const float*)d_in[7];
  const float* bv  = (const float*)d_in[8];
  float* out = (float*)d_out;

  char* ws = (char*)d_ws;
  _Float16* packedW1 = (_Float16*)(ws);                       // 64 KB
  float* xk     = (float*)(ws + (64 << 10));                  // 512 KB
  float* xq     = (float*)(ws + (64 << 10) + (512 << 10));    // 512 KB
  float* vbuf   = (float*)(ws + (64 << 10) + (1024 << 10));   // 256 KB
  float* scores = (float*)(ws + (64 << 10) + (1280 << 10));   // 2 MB

  // Stage A
  {
    int total = BB * TT * CC + BB * TT * HSZ;  // 196608
    stage_a_proj<<<(total + 255) / 256, 256, 0, stream>>>(x, W1, b1, Wv, bv,
                                                          xk, xq, vbuf);
    stage_a_packW1<<<32768 / 256, 256, 0, stream>>>(W1, packedW1);
  }

  // Stage B: grid-stride over 16384 rectangular tile tasks (half are causal)
  stage_b_scores<<<1024, 256, 0, stream>>>(pde, packedW1, xk, xq, W2, b2,
                                           scores);

  // Stage C: one block per (b, i) row
  stage_c_softmax_av<<<BB * TT, 256, 0, stream>>>(scores, vbuf, out);
}
